// SRNN_62843961475311
// MI455X (gfx1250) — matrix operand order repaired and checked by reading, not compiled, not run on hardware
//
#include <hip/hip_runtime.h>

typedef __attribute__((ext_vector_type(16))) __bf16 v16bf;
typedef __attribute__((ext_vector_type(8)))  float  v8f;
typedef __attribute__((ext_vector_type(4)))  unsigned int uint32x4;
typedef __attribute__((ext_vector_type(4)))  int int32x4;
typedef __attribute__((ext_vector_type(8)))  int int32x8;

#if defined(__has_builtin)
#if __has_builtin(__builtin_amdgcn_tensor_load_to_lds) && __has_builtin(__builtin_amdgcn_s_wait_tensorcnt)
#define HAVE_TDM 1
#endif
#endif

// ---------------------------------------------------------------------------
// Libcall-free transcendentals: v_exp_f32 via __builtin_amdgcn_exp2f,
// clamp via v_min_num/v_max_num (no cmp/cndmask chains).
// ---------------------------------------------------------------------------
#define LOG2E 1.4426950408889634f

__device__ __forceinline__ float exp_f(float x) {
  return __builtin_amdgcn_exp2f(x * LOG2E);
}
__device__ __forceinline__ float tanh_f(float x) {
  float xc = __builtin_fmaxf(__builtin_fminf(x, 15.f), -15.f);
  float e2 = __builtin_amdgcn_exp2f(xc * (2.f * LOG2E));
  return (e2 - 1.f) / (e2 + 1.f);
}
__device__ __forceinline__ float sigmoid_f(float x) {
  return 1.f / (1.f + __builtin_amdgcn_exp2f(-x * LOG2E));
}

// ---------------------------------------------------------------------------
// Tensor Data Mover: 2D tile load Global->LDS (D# per CDNA5 ISA).
// data_size=8B elements. Optional LDS padding: after every 128 DWORDs stored,
// skip 64 DWORDs (i.e. land 512B rows on a 768B LDS pitch).
// Issue from one wave; completion via s_wait_tensorcnt.
// ---------------------------------------------------------------------------
#ifdef HAVE_TDM
__device__ __forceinline__ void tdm_load_2d(const void* gaddr, unsigned lds_off,
                                            unsigned tile_d0_e8, unsigned tile_d1,
                                            unsigned stride_e8, int pad_en) {
  unsigned long long ga = (unsigned long long)gaddr;
  uint32x4 g0;
  g0[0] = 1u;                                             // count=1 (valid user D#)
  g0[1] = lds_off;                                        // lds_addr (bytes)
  g0[2] = (unsigned)(ga & 0xFFFFFFFFu);                   // global_addr[31:0]
  g0[3] = (unsigned)((ga >> 32) & 0x01FFFFFFu) | (2u << 30); // addr[56:32] | type=2
  int32x8 g1;
  // data_size=3 (8B) @ [17:16]; pad_enable @ [20]; pad_interval=6 (128 DW) @ [24:22];
  // pad_amount=63 (64 DW) @ [31:25]
  g1[0] = (int)((3u << 16) | (pad_en ? ((1u << 20) | (6u << 22) | (63u << 25)) : 0u));
  g1[1] = (int)((tile_d0_e8 & 0xFFFFu) << 16);            // tensor_dim0[15:0]
  g1[2] = (int)(((tile_d0_e8 >> 16) & 0xFFFFu) | ((tile_d1 & 0xFFFFu) << 16)); // dim0 hi | tensor_dim1
  g1[3] = (int)((tile_d0_e8 & 0xFFFFu) << 16);            // tile_dim0
  g1[4] = (int)(tile_d1 & 0xFFFFu);                       // tile_dim1 (tile_dim2=0)
  g1[5] = (int)stride_e8;                                 // tensor_dim0_stride[31:0]
  g1[6] = 0; g1[7] = 0;
  int32x4 z4 = {0, 0, 0, 0};
  int32x8 z8 = {0, 0, 0, 0, 0, 0, 0, 0};
  __builtin_amdgcn_tensor_load_to_lds(g0, g1, z4, z4, z8, 0);
}
#endif

// ---------------------------------------------------------------------------
// Fragment loaders (CDNA5 wave32 WMMA layouts, see CDNA5 ISA)
// A (16x32 bf16): lane l -> row m=l&15, half h=l>>4; elem j -> K = 8h+j (+8 if j>=8)
// B (32x16 bf16): pre-packed so lane l reads 16 contiguous bf16 (two b128 loads)
// C/D (16x16 f32): vgpr v, lane l -> (M = v + 8h, N = l&15)
// ---------------------------------------------------------------------------

__device__ __forceinline__ v16bf load_a_frag_bf16(const __bf16* A, int lda, int kOff) {
  const int lane = threadIdx.x & 31;
  const int m = lane & 15, h = lane >> 4;
  const __bf16* p = A + (size_t)m * lda + kOff + 8 * h;
  union { uint4 u; __bf16 e[8]; } lo, hi;
  lo.u = *reinterpret_cast<const uint4*>(p);
  hi.u = *reinterpret_cast<const uint4*>(p + 16);
  v16bf a;
#pragma unroll
  for (int j = 0; j < 8; ++j) { a[j] = lo.e[j]; a[j + 8] = hi.e[j]; }
  return a;
}

__device__ __forceinline__ v16bf load_a_frag_f32(const float* A, int lda, int kOff) {
  const int lane = threadIdx.x & 31;
  const int m = lane & 15, h = lane >> 4;
  const float* p = A + (size_t)m * lda + kOff + 8 * h;
  v16bf a;
#pragma unroll
  for (int j = 0; j < 8; ++j) { a[j] = (__bf16)p[j]; a[j + 8] = (__bf16)p[16 + j]; }
  return a;
}

__device__ __forceinline__ v16bf load_b_packed(const __bf16* Wt) {
  const int lane = threadIdx.x & 31;
  const __bf16* p = Wt + lane * 16;
  union { uint4 u; __bf16 e[8]; } u0, u1;
  u0.u = *reinterpret_cast<const uint4*>(p);
  u1.u = *reinterpret_cast<const uint4*>(p + 8);
  v16bf b;
#pragma unroll
  for (int j = 0; j < 8; ++j) { b[j] = u0.e[j]; b[j + 8] = u1.e[j]; }
  return b;
}

// ---------------------------------------------------------------------------
// K0: repack fp32 weight [K,N] -> bf16 B-fragment layout, col-tile-major:
//   packed[((nt*numKT + kt)*32 + lane)*16 + j] = W[kt*32 + 8*(lane>>4) + (j&7) + 16*(j>>3)][nt*16 + (lane&15)]
// ---------------------------------------------------------------------------
__global__ void pack_weight_kernel(const float* __restrict__ W, __bf16* __restrict__ out,
                                   int K, int N) {
  int idx = blockIdx.x * blockDim.x + threadIdx.x;
  if (idx >= K * N) return;
  int j       = idx & 15;
  int lane    = (idx >> 4) & 31;
  int tileIdx = idx >> 9;
  int numKT   = K >> 5;
  int kt = tileIdx % numKT;
  int nt = tileIdx / numKT;
  int k = kt * 32 + ((lane >> 4) << 3) + (j & 7) + ((j >> 3) << 4);
  int n = nt * 16 + (lane & 15);
  out[idx] = (__bf16)W[k * N + n];
}

// ---------------------------------------------------------------------------
// K1/K2/K4: generic WMMA GEMM, one 16x16 tile per wave.
//   out = act(A @ Wpacked + bias0 [+ bias1])
// ---------------------------------------------------------------------------
enum { ACT_NONE = 0, ACT_TANH = 1 };

template <int ACT, bool IN_BF16, bool OUT_BF16>
__global__ void wmma_gemm_kernel(const void* __restrict__ A_, int lda,
                                 const __bf16* __restrict__ Wp,
                                 const float* __restrict__ bias0,
                                 const float* __restrict__ bias1,
                                 void* __restrict__ out_, int ldo,
                                 int M, int Kdim, int NT) {
  const int wavesPerBlock = blockDim.x >> 5;
  const int tile = blockIdx.x * wavesPerBlock + (threadIdx.x >> 5);
  const int totalTiles = (M >> 4) * NT;
  if (tile >= totalTiles) return;
  const int rowT = tile / NT;
  const int colT = tile - rowT * NT;
  const int lane = threadIdx.x & 31;
  const int h = lane >> 4, n = lane & 15;
  const int KT = Kdim >> 5;

  float bias = bias0 ? bias0[colT * 16 + n] : 0.f;
  if (bias1) bias += bias1[colT * 16 + n];
  v8f acc;
#pragma unroll
  for (int v = 0; v < 8; ++v) acc[v] = bias;

  const __bf16* wtile = Wp + ((size_t)(colT * KT) << 9);
  for (int kt = 0; kt < KT; ++kt) {
    v16bf a;
    if (IN_BF16) a = load_a_frag_bf16((const __bf16*)A_ + (size_t)rowT * 16 * lda, lda, kt * 32);
    else         a = load_a_frag_f32 ((const float*)A_  + (size_t)rowT * 16 * lda, lda, kt * 32);
    v16bf b = load_b_packed(wtile + ((size_t)kt << 9));
    acc = __builtin_amdgcn_wmma_f32_16x16x32_bf16(false, a, false, b, (short)0, acc, false, false);
  }
#pragma unroll
  for (int v = 0; v < 8; ++v) {
    float x = acc[v];
    if (ACT == ACT_TANH) x = tanh_f(x);
    const size_t row = (size_t)rowT * 16 + v + 8 * h;
    const size_t idx = row * (size_t)ldo + colT * 16 + n;
    if (OUT_BF16) ((__bf16*)out_)[idx] = (__bf16)x;
    else          ((float*)out_)[idx]  = x;
  }
}

// ---------------------------------------------------------------------------
// K3: RNN scan. 16 blocks x 512 threads; block handles 16 batch rows for all
// 512 steps. W_hh (packed bf16, 128KB) staged to LDS by the TDM; d-state tile
// in LDS.  d_new = tanh(r_t + d @ W_hh);  r_t precomputed (fp32).
// Wave w owns output column tile w (8 WMMA/step). d_seq -> DX[:, 0:256].
// ---------------------------------------------------------------------------
__global__ void rnn_scan_kernel(const float* __restrict__ r,
                                const __bf16* __restrict__ pWhh,
                                __bf16* __restrict__ DX,
                                float* __restrict__ dn_out) {
  extern __shared__ char smem[];
  __bf16* sW = (__bf16*)smem;       // 256x256 packed bf16 = 128KB @ lds_off 0
  __bf16* sD = sW + 65536;          // 16x256 bf16 state tile
  const int b0  = blockIdx.x * 16;
  const int tid = threadIdx.x;

#ifdef HAVE_TDM
  if (tid < 32) {  // one wave drives the DMA; 16384 x 8B contiguous
    tdm_load_2d(pWhh, 0u, 16384u, 1u, 16384u, 0);
    __builtin_amdgcn_s_wait_tensorcnt(0);
  }
#else
  {
    const uint4* src = (const uint4*)pWhh;
    uint4* dst = (uint4*)sW;
    for (int i = tid; i < 8192; i += 512) dst[i] = src[i];
  }
#endif
  for (int i = tid; i < 4096; i += 512) sD[i] = (__bf16)0.f;
  __syncthreads();

  const int wave = tid >> 5;
  const int lane = tid & 31;
  const int h = lane >> 4, n = lane & 15;

  for (int t = 0; t < 512; ++t) {
    const size_t rowBase = (size_t)t * 256 + b0;
    v8f acc;
    const float* rb = r + rowBase * 256 + wave * 16 + n;
#pragma unroll
    for (int v = 0; v < 8; ++v) acc[v] = rb[(size_t)(v + 8 * h) * 256];
#pragma unroll
    for (int kt = 0; kt < 8; ++kt) {
      v16bf a = load_a_frag_bf16(sD, 256, kt * 32);
      v16bf b = load_b_packed(sW + ((size_t)(wave * 8 + kt) << 9));
      acc = __builtin_amdgcn_wmma_f32_16x16x32_bf16(false, a, false, b, (short)0, acc, false, false);
    }
    __syncthreads();   // all reads of old d done
#pragma unroll
    for (int v = 0; v < 8; ++v) {
      float dval = tanh_f(acc[v]);
      int row = v + 8 * h;
      int col = wave * 16 + n;
      __bf16 db = (__bf16)dval;
      sD[row * 256 + col] = db;
      DX[(rowBase + row) * 512 + col] = db;
      if (t == 511) dn_out[(b0 + row) * 256 + col] = dval;
    }
    __syncthreads();
  }
}

// ---------------------------------------------------------------------------
// K5: z scan. 16 blocks x 512 threads; block handles 16 batch rows.
// Per step: X_post=[z|a_t], X_prior=[z|d_t] in LDS (bf16, pitch 384); the TDM
// drops the 16x512B a_t/d_t tiles onto the 768B LDS pitch via descriptor
// padding. Gated hidden (64 col tiles, K=384), heads (24 tiles, K=512),
// fp32 combine + reparameterized sample. DBlock weights stream from L2.
// ---------------------------------------------------------------------------
__global__ void z_scan_kernel(const __bf16* __restrict__ a_seq,
                              const __bf16* __restrict__ DX,
                              const float* __restrict__ eps,
                              const __bf16* pf1p, const __bf16* pf2p,
                              const __bf16* pmup, const __bf16* plsp,
                              const __bf16* pf1q, const __bf16* pf2q,
                              const __bf16* pmuq,
                              const float* bf1p, const float* bf2p,
                              const float* bmup, const float* blsp,
                              const float* bf1q, const float* bf2q,
                              const float* bmuq,
                              float* __restrict__ out_mu,
                              float* __restrict__ out_ls,
                              float* __restrict__ out_zn) {
  extern __shared__ char smem[];
  __bf16* sXp = (__bf16*)smem;        // 16x384 @ lds_off 0
  __bf16* sXq = sXp + 16 * 384;       // 16x384 @ lds_off 12288
  __bf16* sTp = sXq + 16 * 384;       // 16x512
  __bf16* sTq = sTp + 16 * 512;       // 16x512
  float*  sMuP = (float*)(sTq + 16 * 512); // 16x128
  float*  sLs  = sMuP + 2048;
  float*  sMuQ = sLs + 2048;

  const int b0  = blockIdx.x * 16;
  const int tid = threadIdx.x;
  const int wave = tid >> 5;
  const int lane = tid & 31;
  const int h = lane >> 4, n = lane & 15;

  for (int i = tid; i < 16 * 128; i += 512) {   // z0 = 0
    int rr = i >> 7, c = i & 127;
    sXp[rr * 384 + c] = (__bf16)0.f;
    sXq[rr * 384 + c] = (__bf16)0.f;
  }
  __syncthreads();

  for (int t = 0; t < 512; ++t) {
    const size_t rowBase = (size_t)t * 256 + b0;

    // phase 1: stage a_t / d_t tiles into the X concat buffers (cols 128..383)
#ifdef HAVE_TDM
    if (tid < 32) {
      // a_t: 16 rows x 512B contiguous; land at sXp row pitch 768B, col 256B
      tdm_load_2d(a_seq + rowBase * 256, 256u, 64u, 16u, 64u, 1);
      // d_t: 16 rows x 512B with 1024B global stride; land at sXq + 256B
      tdm_load_2d(DX + rowBase * 512, 12288u + 256u, 64u, 16u, 128u, 1);
      __builtin_amdgcn_s_wait_tensorcnt(0);
    }
#else
    for (int i = tid; i < 16 * 256; i += 512) {
      int rr = i >> 8, c = i & 255;
      sXp[rr * 384 + 128 + c] = a_seq[(rowBase + rr) * 256 + c];
      sXq[rr * 384 + 128 + c] = DX[(rowBase + rr) * 512 + c];
    }
#endif
    __syncthreads();

    // phase 2: gated hidden (post: jobs 0..31, prior: 32..63), K=384
    for (int job = wave; job < 64; job += 16) {
      const bool prior = job >= 32;
      const int nt = prior ? job - 32 : job;
      const __bf16* sX = prior ? sXq : sXp;
      const __bf16* W1 = prior ? pf1q : pf1p;
      const __bf16* W2 = prior ? pf2q : pf2p;
      const float*  B1 = prior ? bf1q : bf1p;
      const float*  B2 = prior ? bf2q : bf2p;
      __bf16* sT = prior ? sTq : sTp;
      v8f acc1, acc2;
      const float c1 = B1[nt * 16 + n], c2 = B2[nt * 16 + n];
#pragma unroll
      for (int v = 0; v < 8; ++v) { acc1[v] = c1; acc2[v] = c2; }
      const __bf16* w1t = W1 + ((size_t)(nt * 12) << 9);
      const __bf16* w2t = W2 + ((size_t)(nt * 12) << 9);
#pragma unroll
      for (int kt = 0; kt < 12; ++kt) {
        v16bf a  = load_a_frag_bf16(sX, 384, kt * 32);
        v16bf b1 = load_b_packed(w1t + ((size_t)kt << 9));
        acc1 = __builtin_amdgcn_wmma_f32_16x16x32_bf16(false, a, false, b1, (short)0, acc1, false, false);
        v16bf b2 = load_b_packed(w2t + ((size_t)kt << 9));
        acc2 = __builtin_amdgcn_wmma_f32_16x16x32_bf16(false, a, false, b2, (short)0, acc2, false, false);
        if (kt + 1 < 12) {  // L2-resident weight stream prefetch
          __builtin_prefetch(w1t + ((size_t)(kt + 1) << 9), 0, 0);
          __builtin_prefetch(w2t + ((size_t)(kt + 1) << 9), 0, 0);
        }
      }
#pragma unroll
      for (int v = 0; v < 8; ++v) {
        float tv = tanh_f(acc1[v]) * sigmoid_f(acc2[v]);
        sT[(v + 8 * h) * 512 + nt * 16 + n] = (__bf16)tv;
      }
    }
    __syncthreads();

    // phase 3: heads (post_mu 0..7, post_ls 8..15, prior_mu 16..23), K=512
    for (int job = wave; job < 24; job += 16) {
      const __bf16* sT; const __bf16* W; const float* Bb; float* outL; int nt;
      if (job < 8)       { sT = sTp; W = pmup; Bb = bmup; outL = sMuP; nt = job; }
      else if (job < 16) { sT = sTp; W = plsp; Bb = blsp; outL = sLs;  nt = job - 8; }
      else               { sT = sTq; W = pmuq; Bb = bmuq; outL = sMuQ; nt = job - 16; }
      v8f acc;
      const float cb = Bb[nt * 16 + n];
#pragma unroll
      for (int v = 0; v < 8; ++v) acc[v] = cb;
      const __bf16* wt = W + ((size_t)(nt * 16) << 9);
#pragma unroll
      for (int kt = 0; kt < 16; ++kt) {
        v16bf a = load_a_frag_bf16(sT, 512, kt * 32);
        v16bf b = load_b_packed(wt + ((size_t)kt << 9));
        acc = __builtin_amdgcn_wmma_f32_16x16x32_bf16(false, a, false, b, (short)0, acc, false, false);
      }
#pragma unroll
      for (int v = 0; v < 8; ++v)
        outL[(v + 8 * h) * 128 + nt * 16 + n] = acc[v];
    }
    __syncthreads();

    // phase 4: mu = prior_mu + mean_res; z = mu + exp(0.5*ls)*eps (fp32)
    for (int i = tid; i < 2048; i += 512) {
      int rr = i >> 7, c = i & 127;
      float mu = sMuQ[i] + sMuP[i];
      float ls = sLs[i];
      float z  = mu + exp_f(0.5f * ls) * eps[(rowBase + rr) * 128 + c];
      out_mu[(rowBase + rr) * 128 + c] = mu;
      out_ls[(rowBase + rr) * 128 + c] = ls;
      __bf16 zb = (__bf16)z;
      sXp[rr * 384 + c] = zb;
      sXq[rr * 384 + c] = zb;
      if (t == 511) out_zn[(b0 + rr) * 128 + c] = z;
    }
    __syncthreads();
  }
}

// ---------------------------------------------------------------------------
// Host orchestration
// ---------------------------------------------------------------------------
extern "C" void kernel_launch(void* const* d_in, const int* in_sizes, int n_in,
                              void* d_out, int out_size, void* d_ws, size_t ws_size,
                              hipStream_t stream) {
  (void)in_sizes; (void)n_in; (void)out_size; (void)ws_size;
  const int L = 512, B = 256, K = 256, STATE = 128, IN = 64;
  const int M = L * B;  // 131072 fused (t,b) rows

  const float* ext   = (const float*)d_in[0];
  const float* obs   = (const float*)d_in[1];
  const float* eps   = (const float*)d_in[2];
  const float* Wu    = (const float*)d_in[3];
  const float* bu    = (const float*)d_in[4];
  const float* Wx    = (const float*)d_in[5];
  const float* bx    = (const float*)d_in[6];
  const float* W_ih  = (const float*)d_in[7];
  const float* b_ih  = (const float*)d_in[8];
  const float* W_hh  = (const float*)d_in[9];
  const float* b_hh  = (const float*)d_in[10];
  const float* Wa1   = (const float*)d_in[11];
  const float* ba1   = (const float*)d_in[12];
  const float* Wa2   = (const float*)d_in[13];
  const float* ba2   = (const float*)d_in[14];
  const float* Pf1W  = (const float*)d_in[15]; const float* Pf1b = (const float*)d_in[16];
  const float* Pf2W  = (const float*)d_in[17]; const float* Pf2b = (const float*)d_in[18];
  const float* PmuW  = (const float*)d_in[19]; const float* Pmub = (const float*)d_in[20];
  const float* PlsW  = (const float*)d_in[21]; const float* Plsb = (const float*)d_in[22];
  const float* Qf1W  = (const float*)d_in[23]; const float* Qf1b = (const float*)d_in[24];
  const float* Qf2W  = (const float*)d_in[25]; const float* Qf2b = (const float*)d_in[26];
  const float* QmuW  = (const float*)d_in[27]; const float* Qmub = (const float*)d_in[28];

  char* ws = (char*)d_ws;
  size_t off = 0;
  auto alloc = [&](size_t bytes) -> char* {
    off = (off + 255) & ~(size_t)255;
    char* p = ws + off;
    off += bytes;
    return p;
  };

  __bf16* pWu   = (__bf16*)alloc((size_t)IN * K * 2);
  __bf16* pWx   = (__bf16*)alloc((size_t)IN * K * 2);
  __bf16* pWih  = (__bf16*)alloc((size_t)K * K * 2);
  __bf16* pWhh  = (__bf16*)alloc((size_t)K * K * 2);
  __bf16* pWa1  = (__bf16*)alloc((size_t)2 * K * K * 2);
  __bf16* pWa2  = (__bf16*)alloc((size_t)K * K * 2);
  __bf16* pPf1  = (__bf16*)alloc((size_t)384 * 512 * 2);
  __bf16* pPf2  = (__bf16*)alloc((size_t)384 * 512 * 2);
  __bf16* pPmu  = (__bf16*)alloc((size_t)512 * 128 * 2);
  __bf16* pPls  = (__bf16*)alloc((size_t)512 * 128 * 2);
  __bf16* pQf1  = (__bf16*)alloc((size_t)384 * 512 * 2);
  __bf16* pQf2  = (__bf16*)alloc((size_t)384 * 512 * 2);
  __bf16* pQmu  = (__bf16*)alloc((size_t)512 * 128 * 2);
  __bf16* u_emb = (__bf16*)alloc((size_t)M * K * 2);       // tanh(ext@Wu+bu)
  __bf16* DX    = (__bf16*)alloc((size_t)M * 512 * 2);     // [d_seq | x_emb]
  float*  rbuf  = (float*)alloc((size_t)M * K * 4);        // u_emb@W_ih + b_ih + b_hh
  __bf16* h1    = (__bf16*)alloc((size_t)M * K * 2);       // tanh(DX@Wa1+ba1)
  __bf16* a_seq = (__bf16*)alloc((size_t)M * K * 2);       // h1@Wa2 + ba2

  auto pack = [&](const float* src, __bf16* dst, int Kd, int Nd) {
    int total = Kd * Nd;
    pack_weight_kernel<<<(total + 255) / 256, 256, 0, stream>>>(src, dst, Kd, Nd);
  };
  pack(Wu, pWu, IN, K);     pack(Wx, pWx, IN, K);
  pack(W_ih, pWih, K, K);   pack(W_hh, pWhh, K, K);
  pack(Wa1, pWa1, 2 * K, K); pack(Wa2, pWa2, K, K);
  pack(Pf1W, pPf1, 384, 512); pack(Pf2W, pPf2, 384, 512);
  pack(PmuW, pPmu, 512, 128); pack(PlsW, pPls, 512, 128);
  pack(Qf1W, pQf1, 384, 512); pack(Qf2W, pQf2, 384, 512);
  pack(QmuW, pQmu, 512, 128);

  // Parallel GEMMs: 1 wave per 16x16 tile, 8 waves per block.
  const int grid16 = ((M / 16) * 16 + 7) / 8;  // NT=16 everywhere here
  // u_emb = tanh(ext @ Wu + bu)
  wmma_gemm_kernel<ACT_TANH, false, true><<<grid16, 256, 0, stream>>>(
      ext, IN, pWu, bu, nullptr, u_emb, K, M, IN, 16);
  // x_emb = tanh(obs @ Wx + bx) -> DX columns 256..511
  wmma_gemm_kernel<ACT_TANH, false, true><<<grid16, 256, 0, stream>>>(
      obs, IN, pWx, bx, nullptr, DX + 256, 512, M, IN, 16);
  // r = u_emb @ W_ih + b_ih + b_hh  (hoisted out of the recurrence, fp32)
  wmma_gemm_kernel<ACT_NONE, true, false><<<grid16, 256, 0, stream>>>(
      u_emb, K, pWih, b_ih, b_hh, rbuf, K, M, K, 16);

  float* out    = (float*)d_out;
  float* out_mu = out;
  float* out_ls = out + (size_t)M * STATE;
  float* out_dn = out + 2 * (size_t)M * STATE;
  float* out_zn = out_dn + (size_t)B * K;

  // RNN scan: 16 persistent blocks (16 batch rows each), W_hh in LDS via TDM.
  rnn_scan_kernel<<<16, 512, 65536 * 2 + 4096 * 2, stream>>>(rbuf, pWhh, DX, out_dn);

  // a_seq = tanh([d|x] @ Wa1 + ba1) @ Wa2 + ba2
  wmma_gemm_kernel<ACT_TANH, true, true><<<grid16, 256, 0, stream>>>(
      DX, 512, pWa1, ba1, nullptr, h1, K, M, 512, 16);
  wmma_gemm_kernel<ACT_NONE, true, true><<<grid16, 256, 0, stream>>>(
      h1, K, pWa2, ba2, nullptr, a_seq, K, M, K, 16);

  // z scan: 16 persistent blocks; DBlock weights stream from L2, a_t/d_t
  // tiles staged by the TDM with LDS pitch padding.
  const size_t zsh = (size_t)2 * 16 * 384 * 2 + 2 * 16 * 512 * 2 + 3 * 2048 * 4;
  z_scan_kernel<<<16, 512, zsh, stream>>>(
      a_seq, DX, eps,
      pPf1, pPf2, pPmu, pPls, pQf1, pQf2, pQmu,
      Pf1b, Pf2b, Pmub, Plsb, Qf1b, Qf2b, Qmub,
      out_mu, out_ls, out_zn);
}
